// graph_cp_pooling_50818053046958
// MI455X (gfx1250) — matrix operand order repaired and checked by reading, not compiled, not run on hardware
//
#include <hip/hip_runtime.h>

// ---------- types ----------
typedef __bf16 v16bf __attribute__((ext_vector_type(16)));
typedef __bf16 v8bf  __attribute__((ext_vector_type(8)));
typedef float  v8f   __attribute__((ext_vector_type(8)));
typedef unsigned int uint32x4 __attribute__((ext_vector_type(4)));
typedef int int32x4 __attribute__((ext_vector_type(4)));
typedef int int32x8 __attribute__((ext_vector_type(8)));

#define IN_FEA_C 512          // K dimension (compile-time)
#define RCHUNK   128          // ranks handled per block
#define PITCH    520          // LDS row pitch in bf16 elems (1040B: 4-bank rotate/row, 16B aligned)
#define NTILE    16           // nodes per M tile

// split fp32 -> bf16 hi (truncate) + bf16 lo (round residual); hi+lo ~ 2^-15 rel err
__device__ __forceinline__ void split_f32(float x, unsigned short& h, unsigned short& l) {
  unsigned u = __builtin_bit_cast(unsigned, x);
  h = (unsigned short)(u >> 16);
  float hf = __builtin_bit_cast(float, u & 0xFFFF0000u);
  float r  = x - hf;                       // exact
  unsigned ru = __builtin_bit_cast(unsigned, r);
  ru += 0x7FFFu + ((ru >> 16) & 1u);       // RN to bf16
  l = (unsigned short)(ru >> 16);
}

// ---------- kernel 1: split W once into bf16 hi/lo planes in workspace ----------
__global__ void wsplit_kernel(const float* __restrict__ W,
                              unsigned short* __restrict__ dst, int total) {
  int i = blockIdx.x * blockDim.x + threadIdx.x;
  if (i >= total) return;
  unsigned short h, l;
  split_f32(W[i], h, l);
  dst[i]         = h;   // plane 0: hi
  dst[total + i] = l;   // plane 1: lo
}

// ---------- TDM: DMA one 128-row x 1024B bf16 plane into LDS with 16B/row padding ----------
// Tile maps exactly onto the D# of CDNA5 ISA: data_size=8B, dims 128x128 units,
// dim0_stride=128, pad_interval=7 (256 DWORDs = 1024B), pad_amount=3 (4 DWORDs = 16B).
__device__ __forceinline__ void tdm_load_w_plane(const unsigned short* src, unsigned ldsOff) {
#if __has_builtin(__builtin_amdgcn_tensor_load_to_lds)
  unsigned long long ga = (unsigned long long)(size_t)(const void*)src;
  uint32x4 g0 = { 1u,                                   // count=1 (valid user descriptor)
                  ldsOff,                               // lds_addr (bytes)
                  (unsigned)(ga & 0xFFFFFFFFu),         // global_addr[31:0]
                  (unsigned)((ga >> 32) & 0x01FFFFFFu) | (2u << 30) };  // [56:32] | type=2
  int32x8 g1 = { (int)((3u << 16) | (1u << 20) | (7u << 22) | (3u << 25)),
                 (int)(128u << 16),    // tensor_dim0[15:0]=128  (8B units)
                 (int)(128u << 16),    // tensor_dim1[15:0]=128  (bit 80 = w2[16])
                 (int)(128u << 16),    // tile_dim0=128          (bits 127:112)
                 (int)(128u),          // tile_dim1=128, tile_dim2=0
                 (int)(128u),          // tensor_dim0_stride[31:0]=128
                 0, 0 };               // stride hi / tensor_dim1_stride = 0
  int32x4 gz = { 0, 0, 0, 0 };
#if __clang_major__ >= 23
  int32x8 gz8 = { 0, 0, 0, 0, 0, 0, 0, 0 };
  __builtin_amdgcn_tensor_load_to_lds(g0, g1, gz, gz, gz8, 0);
#else
  __builtin_amdgcn_tensor_load_to_lds(g0, g1, gz, gz, 0);
#endif
#else
  (void)src; (void)ldsOff;
#endif
}

// ---------- kernel 2: fused projection + tanh + segmented product ----------
__launch_bounds__(256, 1)
__global__ void cp_pool_kernel(const float* __restrict__ x,
                               const unsigned short* __restrict__ whl,  // [2][rank][512] bf16 bits
                               const float* __restrict__ bias,
                               const int* __restrict__ ids,
                               float* __restrict__ out,
                               int nNodes, int rank) {
  __shared__ __align__(128) unsigned short sW[2][RCHUNK][PITCH];  // 266,240 B
  __shared__ __align__(128) unsigned short sX[2][NTILE][PITCH];   //  33,280 B (total 299,520 <= 320K)

  const int tid  = threadIdx.x;
  const int lane = tid & 31;
  const int wv   = tid >> 5;                 // wave id 0..7; wave owns 16 ranks
  const int chunks = rank >> 7;              // rank / 128
  const int seg  = blockIdx.x / chunks;
  const int half = blockIdx.x - seg * chunks;
  const int rbase = half * RCHUNK;

  // uniform binary searches over sorted segment_ids -> [segStart, segEnd)
  int lo = 0, hi = nNodes;
  while (lo < hi) { int mid = (lo + hi) >> 1; if (ids[mid] < seg) lo = mid + 1; else hi = mid; }
  const int segStart = lo;
  hi = nNodes;
  while (lo < hi) { int mid = (lo + hi) >> 1; if (ids[mid] < seg + 1) lo = mid + 1; else hi = mid; }
  const int segEnd = lo;

  // ---- stage W hi/lo chunk (this block's 128 ranks) into LDS ----
#if __has_builtin(__builtin_amdgcn_tensor_load_to_lds)
  if (wv == 0) {
    tdm_load_w_plane(whl + (size_t)rbase * IN_FEA_C,
                     (unsigned)(size_t)(void*)&sW[0][0][0]);
    tdm_load_w_plane(whl + ((size_t)rank + rbase) * IN_FEA_C,
                     (unsigned)(size_t)(void*)&sW[1][0][0]);
  }
  __builtin_amdgcn_s_wait_tensorcnt((short)0);
#else
  {
    const int nChunk = 2 * RCHUNK * (IN_FEA_C / 8);     // 16384 16B granules
    for (int i = tid; i < nChunk; i += 256) {
      int plane = i >> 13;
      int rem   = i & 8191;
      int r     = rem >> 6;
      int c8    = rem & 63;
      const uint4 v = ((const uint4*)(whl + ((size_t)plane * rank + rbase + r) * IN_FEA_C))[c8];
      *((uint4*)&sW[plane][r][c8 * 8]) = v;
    }
  }
#endif

  const int n16 = lane & 15;                  // N index within the wave's 16-rank slice
  const int sel = lane >> 4;                  // which K-half this lane holds (ISA layout)
  const float bval = bias[rbase + wv * 16 + n16];
  const unsigned short* wrow_h = &sW[0][wv * 16 + n16][0];
  const unsigned short* wrow_l = &sW[1][wv * 16 + n16][0];
  const unsigned short* xrow_h = &sX[0][n16][0];
  const unsigned short* xrow_l = &sX[1][n16][0];

  float prod = 1.0f;
  float4 reg[8];                              // software-pipelined x tile (8 float4 / thread)

  // prologue: issue loads for first tile (latency overlaps binary search / staging wait)
  if (segStart < segEnd) {
    #pragma unroll
    for (int it = 0; it < 8; ++it) {
      int idx = tid + it * 256;
      int r = idx >> 7, c4 = idx & 127;
      int node = segStart + r;
      float4 v = make_float4(0.f, 0.f, 0.f, 0.f);
      if (node < segEnd) v = ((const float4*)(x + (size_t)node * IN_FEA_C))[c4];
      reg[it] = v;
    }
  }
  __syncthreads();                            // W staged (TDM waited above), sX writable

  for (int base = segStart; base < segEnd; base += NTILE) {
    // ---- Phase A: convert pipelined registers -> LDS bf16 hi/lo tile ----
    #pragma unroll
    for (int it = 0; it < 8; ++it) {
      int idx = tid + it * 256;
      int r = idx >> 7, c4 = idx & 127;
      float4 v = reg[it];
      unsigned short h0, h1, h2, h3, l0, l1, l2, l3;
      split_f32(v.x, h0, l0); split_f32(v.y, h1, l1);
      split_f32(v.z, h2, l2); split_f32(v.w, h3, l3);
      *((ushort4*)&sX[0][r][c4 * 4]) = make_ushort4(h0, h1, h2, h3);
      *((ushort4*)&sX[1][r][c4 * 4]) = make_ushort4(l0, l1, l2, l3);
    }
    __syncthreads();

    // issue next tile's global loads now; their latency hides under the WMMA phase
    {
      int nbase = base + NTILE;
      if (nbase < segEnd) {
        #pragma unroll
        for (int it = 0; it < 8; ++it) {
          int idx = tid + it * 256;
          int r = idx >> 7, c4 = idx & 127;
          int node = nbase + r;
          float4 v = make_float4(0.f, 0.f, 0.f, 0.f);
          if (node < segEnd) v = ((const float4*)(x + (size_t)node * IN_FEA_C))[c4];
          reg[it] = v;
        }
      }
    }

    // ---- Phase B: 16x16 tile GEMM over K=512 with bf16x2 split (3 WMMAs / k-step) ----
    v8f c = {};
    #pragma unroll
    for (int kk = 0; kk < IN_FEA_C / 32; ++kk) {
      // A (16x32 bf16): lane<16 holds K={0..7,16..23}, lane>=16 holds K={8..15,24..31}
      const int aoff = kk * 32 + sel * 8;
      v8bf ah0 = *(const v8bf*)(xrow_h + aoff);
      v8bf ah1 = *(const v8bf*)(xrow_h + aoff + 16);
      v8bf al0 = *(const v8bf*)(xrow_l + aoff);
      v8bf al1 = *(const v8bf*)(xrow_l + aoff + 16);
      // B (32x16 bf16): lane holds column n, lane<16 K={0..7,16..23}, lane>=16 K={8..15,24..31}
      const int boff = kk * 32 + sel * 8;
      v8bf bh0 = *(const v8bf*)(wrow_h + boff);
      v8bf bh1 = *(const v8bf*)(wrow_h + boff + 16);
      v8bf bl0 = *(const v8bf*)(wrow_l + boff);
      v8bf bl1 = *(const v8bf*)(wrow_l + boff + 16);
      v16bf A_h = __builtin_shufflevector(ah0, ah1, 0,1,2,3,4,5,6,7,8,9,10,11,12,13,14,15);
      v16bf A_l = __builtin_shufflevector(al0, al1, 0,1,2,3,4,5,6,7,8,9,10,11,12,13,14,15);
      v16bf B_h = __builtin_shufflevector(bh0, bh1, 0,1,2,3,4,5,6,7,8,9,10,11,12,13,14,15);
      v16bf B_l = __builtin_shufflevector(bl0, bl1, 0,1,2,3,4,5,6,7,8,9,10,11,12,13,14,15);
      c = __builtin_amdgcn_wmma_f32_16x16x32_bf16(false, A_h, false, B_h, (short)0, c, false, false);
      c = __builtin_amdgcn_wmma_f32_16x16x32_bf16(false, A_h, false, B_l, (short)0, c, false, false);
      c = __builtin_amdgcn_wmma_f32_16x16x32_bf16(false, A_l, false, B_h, (short)0, c, false, false);
    }

    // ---- epilogue: bias + tanh + masked multiplicative accumulation ----
    const int vm = segEnd - base;           // valid rows in this tile
    #pragma unroll
    for (int j = 0; j < 8; ++j) {           // C VGPR j: M=j (lanes<16) / M=j+8 (lanes>=16)
      int mrow = j + (sel ? 8 : 0);
      float arg = c[j] + bval;
      float e = __expf(2.0f * arg);         // tanh(a) = 1 - 2/(e^{2a}+1), saturates correctly
      float t = 1.0f - 2.0f / (e + 1.0f);
      prod *= (mrow < vm) ? t : 1.0f;
    }
    __syncthreads();
  }

  // combine the two M-halves held by lane and lane^16, then write 16 ranks per wave
  prod *= __shfl_xor(prod, 16, 32);
  if (lane < 16) out[(size_t)seg * rank + rbase + wv * 16 + lane] = prod;
}

// ---------- host entry ----------
extern "C" void kernel_launch(void* const* d_in, const int* in_sizes, int n_in,
                              void* d_out, int out_size, void* d_ws, size_t ws_size,
                              hipStream_t stream) {
  const float* x  = (const float*)d_in[0];
  const float* W  = (const float*)d_in[1];
  const float* b  = (const float*)d_in[2];
  const int* ids  = (const int*)d_in[3];
  (void)n_in;

  const int rank   = in_sizes[2];                    // 256
  const int inFea  = in_sizes[1] / rank;             // 512
  const int nNodes = in_sizes[0] / inFea;            // 131072
  const int numSeg = out_size / rank;                // 1024
  if (inFea != IN_FEA_C || (rank & (RCHUNK - 1)) != 0) return;

  const size_t wTotal = (size_t)rank * inFea;
  if (ws_size < wTotal * 2 * sizeof(unsigned short)) return;
  unsigned short* whl = (unsigned short*)d_ws;

  wsplit_kernel<<<(unsigned)((wTotal + 255) / 256), 256, 0, stream>>>(W, whl, (int)wTotal);

  dim3 grid((unsigned)(numSeg * (rank / RCHUNK)));   // 2048 blocks: (segment, rank-half)
  cp_pool_kernel<<<grid, 256, 0, stream>>>(x, whl, b, ids, (float*)d_out, nNodes, rank);
}
